// NormalPooling_40845138985512
// MI455X (gfx1250) — matrix operand order repaired and checked by reading, not compiled, not run on hardware
//
#include <hip/hip_runtime.h>
#include <hip/hip_bf16.h>
#include <math.h>

#define S_SEG 128
#define F_DIM 256
#define H_DIM 128
#define LDS_STRIDE 264   // 256 + 8 bf16 pad -> 528B row pitch, de-conflicts LDS banks
#define MAX_LEN 3072

typedef __attribute__((ext_vector_type(16))) __bf16 v16bf;
typedef __attribute__((ext_vector_type(8)))  __bf16 v8bf;
typedef __attribute__((ext_vector_type(8)))  float  v8f;
typedef __attribute__((ext_vector_type(4)))  float  f32x4;

__device__ __forceinline__ __bf16 f2bf(float f) { return (__bf16)f; }

__device__ __forceinline__ v16bf pack16(f32x4 q0, f32x4 q1, f32x4 q2, f32x4 q3) {
    v16bf a;
    a[0]=f2bf(q0.x);  a[1]=f2bf(q0.y);  a[2]=f2bf(q0.z);  a[3]=f2bf(q0.w);
    a[4]=f2bf(q1.x);  a[5]=f2bf(q1.y);  a[6]=f2bf(q1.z);  a[7]=f2bf(q1.w);
    a[8]=f2bf(q2.x);  a[9]=f2bf(q2.y);  a[10]=f2bf(q2.z); a[11]=f2bf(q2.w);
    a[12]=f2bf(q3.x); a[13]=f2bf(q3.y); a[14]=f2bf(q3.z); a[15]=f2bf(q3.w);
    return a;
}

// ---------------- Kernel 1: prefix sum of segment lengths -> offsets[0..128]
__global__ __launch_bounds__(S_SEG) void scan_kernel(const int* __restrict__ lengths,
                                                     int* __restrict__ offsets) {
    __shared__ int buf[2][S_SEG];
    const int tid = threadIdx.x;
    buf[0][tid] = lengths[tid];
    __syncthreads();
    int src = 0;
    for (int off = 1; off < S_SEG; off <<= 1) {
        int t = buf[src][tid];
        if (tid >= off) t += buf[src][tid - off];
        buf[src ^ 1][tid] = t;
        src ^= 1;
        __syncthreads();
    }
    offsets[tid + 1] = buf[src][tid];
    if (tid == 0) offsets[0] = 0;
}

// ---------------- Kernel 2: W1 (256x128 f32, row-major) -> W1^T (128x256 bf16)
__global__ __launch_bounds__(256) void convert_w1_kernel(const float* __restrict__ W1,
                                                         __bf16* __restrict__ w1t) {
    const int i = blockIdx.x * 256 + threadIdx.x;   // 0..32767
    const int n = i >> 8;   // output row   (col of W1), 0..127
    const int k = i & 255;  // output col   (row of W1), 0..255
    w1t[i] = f2bf(W1[k * H_DIM + n]);
}

// ---------------- Kernel 3: y,z = tanh(x@W1+b1)@W2 + b2  via bf16 WMMA
// 8 waves/block, each wave owns a 32-row x 128-col tile (2 A-fragments per
// k-step, each B fragment reused for 2 WMMAs -> halves LDS traffic per FLOP).
__global__ __launch_bounds__(256) void gemm_yz_kernel(
    const float* __restrict__ x, const __bf16* __restrict__ w1t,
    const float* __restrict__ b1, const float* __restrict__ W2,
    const float* __restrict__ b2,
    float* __restrict__ yv, float* __restrict__ zv, int N)
{
    __shared__ __bf16 lds_b[H_DIM * LDS_STRIDE];

    // Stage W1^T into padded LDS (16B vectors), shared by all 8 waves.
    for (int i = threadIdx.x; i < (H_DIM * F_DIM) / 8; i += 256) {
        const int n  = i >> 5;
        const int kv = (i & 31) << 3;
        *(v8bf*)(&lds_b[n * LDS_STRIDE + kv]) = ((const v8bf*)w1t)[i];
    }
    __syncthreads();

    const int wave = threadIdx.x >> 5;
    const int lane = threadIdx.x & 31;
    const int m    = lane & 15;     // row within A tile / col within B,C tile
    const int hi   = lane >> 4;     // lane half
    const int row0 = blockIdx.x * 256 + wave * 32;

    int rowa0 = row0 + m;       if (rowa0 > N - 1) rowa0 = N - 1;
    int rowa1 = row0 + 16 + m;  if (rowa1 > N - 1) rowa1 = N - 1;
    const f32x4* xr0 = (const f32x4*)(x + (size_t)rowa0 * F_DIM);
    const f32x4* xr1 = (const f32x4*)(x + (size_t)rowa1 * F_DIM);

    v8f acc0[8] = {};    // rows row0..row0+15
    v8f acc1[8] = {};    // rows row0+16..row0+31

    #pragma unroll
    for (int s = 0; s < 8; ++s) {             // K = 256 in steps of 32
        const int kq = (s * 32 + hi * 8) >> 2;  // A layout: lanes<16 K[0..7],K[16..23]; +8 for hi
        const f32x4 p0 = __builtin_nontemporal_load(&xr0[kq]);
        const f32x4 p1 = __builtin_nontemporal_load(&xr0[kq + 1]);
        const f32x4 p2 = __builtin_nontemporal_load(&xr0[kq + 4]);
        const f32x4 p3 = __builtin_nontemporal_load(&xr0[kq + 5]);
        const f32x4 q0 = __builtin_nontemporal_load(&xr1[kq]);
        const f32x4 q1 = __builtin_nontemporal_load(&xr1[kq + 1]);
        const f32x4 q2 = __builtin_nontemporal_load(&xr1[kq + 4]);
        const f32x4 q3 = __builtin_nontemporal_load(&xr1[kq + 5]);
        const v16bf a0 = pack16(p0, p1, p2, p3);
        const v16bf a1 = pack16(q0, q1, q2, q3);

        #pragma unroll
        for (int t = 0; t < 8; ++t) {         // 8 col-tiles cover H=128
            // B layout: lane -> col n, elements 0..7 -> K[hi*8 .. hi*8+7], 8..15 -> K[16+hi*8 .. 16+hi*8+7]
            const int boff = (t * 16 + m) * LDS_STRIDE + s * 32 + hi * 8;
            union { v16bf v; v8bf h[2]; } bu;
            bu.h[0] = *(const v8bf*)(&lds_b[boff]);
            bu.h[1] = *(const v8bf*)(&lds_b[boff + 16]);
            acc0[t] = __builtin_amdgcn_wmma_f32_16x16x32_bf16(
                false, a0, false, bu.v, (short)0, acc0[t], false, false);
            acc1[t] = __builtin_amdgcn_wmma_f32_16x16x32_bf16(
                false, a1, false, bu.v, (short)0, acc1[t], false, false);
        }
    }

    // Fused epilogue: h = tanh(acc + b1); y = h@W2[:,0]+b2[0]; z = h@W2[:,1]+b2[1]
    float b1v[8], w20[8], w21[8];
    #pragma unroll
    for (int t = 0; t < 8; ++t) {
        const int n = t * 16 + m;
        b1v[t] = b1[n];
        w20[t] = W2[2 * n];
        w21[t] = W2[2 * n + 1];
    }
    const float b20 = b2[0], b21 = b2[1];

    auto epilogue = [&](const v8f (&acc)[8], int abase) {
        #pragma unroll
        for (int r = 0; r < 8; ++r) {         // C layout: vgpr r -> row r + 8*hi
            float yr = 0.f, zr = 0.f;
            #pragma unroll
            for (int t = 0; t < 8; ++t) {
                const float h = tanhf(acc[t][r] + b1v[t]);
                yr += h * w20[t];
                zr += h * w21[t];
            }
            #pragma unroll
            for (int msk = 1; msk < 16; msk <<= 1) {   // reduce across 16 cols (half-wave)
                yr += __shfl_xor(yr, msk, 32);
                zr += __shfl_xor(zr, msk, 32);
            }
            const int rowg = row0 + abase + hi * 8 + r;
            if (m == r && rowg < N) {
                yv[rowg] = yr + b20;
                zv[rowg] = zr + b21;
            }
        }
    };
    epilogue(acc0, 0);
    epilogue(acc1, 16);
}

// ---------------- Kernel 4: per-segment softmax stats + pdf normalizer
__global__ __launch_bounds__(256) void seg_stats_kernel(
    const float* __restrict__ yv, const float* __restrict__ zv,
    const int* __restrict__ lengths, const int* __restrict__ offsets,
    float* __restrict__ mu, float* __restrict__ sd, float* __restrict__ ps)
{
    __shared__ float red[256];
    const int s     = blockIdx.x;
    const int start = offsets[s];
    const int len   = lengths[s];
    const float Lf  = (float)len;
    const int tid   = threadIdx.x;

    // pass 1: segment max of y
    float mloc = -3.402823466e38f;
    for (int i = tid; i < len; i += 256) mloc = fmaxf(mloc, yv[start + i]);
    red[tid] = mloc; __syncthreads();
    for (int o = 128; o > 0; o >>= 1) { if (tid < o) red[tid] = fmaxf(red[tid], red[tid + o]); __syncthreads(); }
    const float ymax = red[0];
    __syncthreads();

    // pass 2: wsum, sum(xpos*w), sum(z)
    float wsum = 0.f, xw = 0.f, zs = 0.f;
    for (int i = tid; i < len; i += 256) {
        const float w    = expf(yv[start + i] - ymax);
        const float xpos = (float)(i + 1) / Lf;
        wsum += w; xw += w * xpos; zs += zv[start + i];
    }
    red[tid] = wsum; __syncthreads();
    for (int o = 128; o > 0; o >>= 1) { if (tid < o) red[tid] += red[tid + o]; __syncthreads(); }
    wsum = red[0]; __syncthreads();
    red[tid] = xw; __syncthreads();
    for (int o = 128; o > 0; o >>= 1) { if (tid < o) red[tid] += red[tid + o]; __syncthreads(); }
    xw = red[0]; __syncthreads();
    red[tid] = zs; __syncthreads();
    for (int o = 128; o > 0; o >>= 1) { if (tid < o) red[tid] += red[tid + o]; __syncthreads(); }
    zs = red[0]; __syncthreads();

    const float muv = xw / wsum;
    const float v   = zs / Lf;
    const float sdv = (v > 20.f) ? v : log1pf(expf(v));   // softplus

    // pass 3: sum of exp(-0.5*((xpos-mu)/sd)^2)
    const float inv_sd = 1.f / sdv;
    float p = 0.f;
    for (int i = tid; i < len; i += 256) {
        const float xpos = (float)(i + 1) / Lf;
        const float d    = (xpos - muv) * inv_sd;
        p += expf(-0.5f * d * d);
    }
    red[tid] = p; __syncthreads();
    for (int o = 128; o > 0; o >>= 1) { if (tid < o) red[tid] += red[tid + o]; __syncthreads(); }

    if (tid == 0) {
        const float inv_sqrt_2pi = 0.3989422804014327f;
        mu[s] = muv;
        sd[s] = sdv;
        ps[s] = red[0] * inv_sqrt_2pi * inv_sd;   // psum with pdf prefactor folded in
    }
}

// ---------------- Kernel 5: attn per row + pooled output per segment
__global__ __launch_bounds__(256) void seg_out_kernel(
    const float* __restrict__ x, const int* __restrict__ lengths,
    const int* __restrict__ offsets, const float* __restrict__ mu,
    const float* __restrict__ sd, const float* __restrict__ ps,
    float* __restrict__ out, float* __restrict__ attn)
{
    __shared__ float a_lds[MAX_LEN];
    const int s     = blockIdx.x;
    const int start = offsets[s];
    const int len   = lengths[s];
    const float Lf  = (float)len;
    const int tid   = threadIdx.x;

    const float muv    = mu[s];
    const float inv_sd = 1.f / sd[s];
    const float inv_sqrt_2pi = 0.3989422804014327f;
    const float coef = inv_sqrt_2pi * inv_sd / (ps[s] + 0.001f);

    for (int i = tid; i < len; i += 256) {
        const float xpos = (float)(i + 1) / Lf;
        const float d    = (xpos - muv) * inv_sd;
        const float a    = expf(-0.5f * d * d) * coef;
        a_lds[i] = a;
        attn[start + i] = a;
    }
    __syncthreads();

    // thread tid owns feature f = tid; coalesced 1KB row reads across the block
    const float* xb = x + (size_t)start * F_DIM + tid;
    float a0 = 0.f, a1 = 0.f, a2 = 0.f, a3 = 0.f;
    int i = 0;
    for (; i + 4 <= len; i += 4) {
        a0 += a_lds[i]     * __builtin_nontemporal_load(&xb[(size_t)(i)     * F_DIM]);
        a1 += a_lds[i + 1] * __builtin_nontemporal_load(&xb[(size_t)(i + 1) * F_DIM]);
        a2 += a_lds[i + 2] * __builtin_nontemporal_load(&xb[(size_t)(i + 2) * F_DIM]);
        a3 += a_lds[i + 3] * __builtin_nontemporal_load(&xb[(size_t)(i + 3) * F_DIM]);
    }
    for (; i < len; ++i) a0 += a_lds[i] * __builtin_nontemporal_load(&xb[(size_t)i * F_DIM]);
    out[s * F_DIM + tid] = (a0 + a1) + (a2 + a3);
}

// ---------------- host-side orchestration
extern "C" void kernel_launch(void* const* d_in, const int* in_sizes, int n_in,
                              void* d_out, int out_size, void* d_ws, size_t ws_size,
                              hipStream_t stream) {
    const float* x       = (const float*)d_in[0];
    const int*   lengths = (const int*)  d_in[1];
    const float* W1      = (const float*)d_in[2];
    const float* b1      = (const float*)d_in[3];
    const float* W2      = (const float*)d_in[4];
    const float* b2      = (const float*)d_in[5];
    const int N = in_sizes[0] / F_DIM;

    // workspace layout
    char* ws = (char*)d_ws;
    int*    offsets = (int*)ws;                         // 129 ints (<1KB)
    __bf16* w1t     = (__bf16*)(ws + 1024);             // 128*256 bf16 = 64KB
    float*  yv      = (float*)(ws + 1024 + 65536);      // N floats
    float*  zv      = yv + N;                           // N floats
    float*  muv     = zv + N;                           // 128
    float*  sdv     = muv + S_SEG;                      // 128
    float*  psv     = sdv + S_SEG;                      // 128

    float* out  = (float*)d_out;        // 128*256 pooled output
    float* attn = out + S_SEG * F_DIM;  // N attention weights

    hipLaunchKernelGGL(scan_kernel, dim3(1), dim3(S_SEG), 0, stream, lengths, offsets);
    hipLaunchKernelGGL(convert_w1_kernel, dim3((H_DIM * F_DIM) / 256), dim3(256), 0, stream,
                       W1, w1t);
    const int gblocks = (N + 255) / 256;
    hipLaunchKernelGGL(gemm_yz_kernel, dim3(gblocks), dim3(256), 0, stream,
                       x, w1t, b1, W2, b2, yv, zv, N);
    hipLaunchKernelGGL(seg_stats_kernel, dim3(S_SEG), dim3(256), 0, stream,
                       yv, zv, lengths, offsets, muv, sdv, psv);
    hipLaunchKernelGGL(seg_out_kernel, dim3(S_SEG), dim3(256), 0, stream,
                       x, lengths, offsets, muv, sdv, psv, out, attn);
}
